// EvoBinarizedLayer_58780922413282
// MI455X (gfx1250) — compile-verified, matrix operand order checked by reading, not run on hardware
//
#include <hip/hip_runtime.h>

// Problem: P=32, B=512, I=1024, O=1024
//   out[p,b,o] = sum_i x*w0 + (1-x)*w1 = x@(w0-w1) + colsum(w1)
// Single bf16 WMMA GEMM per p; (w0-w1) conversion and bias fused into the
// weight streaming. Memory-bound (~384 MB compulsory @ 23.3 TB/s), so the
// weight stream (256 MB) uses b128 loads exclusively.

typedef __attribute__((ext_vector_type(16))) __bf16 v16bf;
typedef __attribute__((ext_vector_type(8)))  float  v8f;
typedef __attribute__((ext_vector_type(4)))  float  f4;

#define PP 32
#define BB 512
#define II 1024
#define OO 1024

#define BM 128
#define BN 128
#define BK 32
#define KSTEPS (II / BK)   // 32
#define MTILES (BB / BM)   // 4
#define NTILES (OO / BN)   // 8

// exact for {0,1} / {-1,0,1}: plain truncation fp32 -> bf16
static __device__ __forceinline__ unsigned int pack_bf16(float lo, float hi) {
  unsigned int ulo = __builtin_bit_cast(unsigned int, lo) >> 16;
  unsigned int uhi = __builtin_bit_cast(unsigned int, hi) & 0xffff0000u;
  return uhi | ulo;
}

// Fragment-slot math per CDNA5 ISA 16-bit A/B layouts (7.12.2):
//   lane = (row & 15) | (((k >> 3) & 1) << 4)
//   vgpr = ((k >> 4) << 2) | ((k & 7) >> 1)      (k pair packed lo/hi)
static __device__ __forceinline__ int frag_lane(int row, int k) {
  return (row & 15) | (((k >> 3) & 1) << 4);
}
static __device__ __forceinline__ int frag_vgpr(int k) {
  return ((k >> 4) << 2) | ((k & 7) >> 1);
}

__global__ __launch_bounds__(256)
void evo_binarized_wmma(const float* __restrict__ x,
                        const float* __restrict__ w,
                        float* __restrict__ out) {
  const int tile  = blockIdx.x;
  const int mtile = tile & (MTILES - 1);
  const int ntile = tile >> 2;          // MTILES == 4
  const int p     = blockIdx.y;

  const int tid  = threadIdx.x;         // 0..255, 8 wave32 waves
  const int lane = tid & 31;
  const int wave = tid >> 5;
  const int wm   = wave >> 1;           // 0..3  -> 2 M-subtiles each
  const int wn   = wave & 1;            // 0..1  -> 4 N-subtiles each

  const float* xp = x + ((size_t)p * BB + (size_t)mtile * BM) * II;
  const float* w0 = w + (size_t)p        * II * OO + (size_t)ntile * BN;
  const float* w1 = w + (size_t)(PP + p) * II * OO + (size_t)ntile * BN;
  float*     outp = out + ((size_t)p * BB + (size_t)mtile * BM) * OO
                        + (size_t)ntile * BN;

  // LDS tiles in per-lane WMMA fragment layout: [subtile][lane][vgpr]
  __shared__ unsigned int lds_a[8 * 32 * 8];   // 8 KB : A (x) bf16
  __shared__ unsigned int lds_w[8 * 32 * 8];   // 8 KB : B (w0-w1) bf16
  __shared__ float        lds_bp[32 * 16 * 4]; // 8 KB : bias partials
  __shared__ float        lds_bias[BN];        // reduced colsum(w1)

  v8f acc[2][4];
#pragma unroll
  for (int mi = 0; mi < 2; ++mi)
#pragma unroll
    for (int ni = 0; ni < 4; ++ni)
      acc[mi][ni] = v8f{};

  // Weight loader mapping: work item = (kpair, 4-column group).
  //   sub = kpair (0..15), ng0 / ng0+16 = the two column groups this thread owns
  const int sub = tid & 15;
  const int ng0 = tid >> 4;             // 0..15
  f4 bias_acc[2] = {f4{}, f4{}};

  for (int kk = 0; kk < KSTEPS; ++kk) {
    const int k0 = kk * BK;

    // ---- load A: 128x32 fp32, coalesced float4 rows, convert to bf16 ----
#pragma unroll
    for (int pass = 0; pass < 4; ++pass) {
      const int idx = pass * 256 + tid;
      const int m   = idx >> 3;               // 0..127
      const int kg  = (idx & 7) * 4;          // 0,4,...,28
      const f4 v = *(const f4*)(xp + (size_t)m * II + (k0 + kg));
      const unsigned int p0 = pack_bf16(v[0], v[1]);
      const unsigned int p1 = pack_bf16(v[2], v[3]);
      const int mt = m >> 4;
      const int L  = frag_lane(m, kg);
      const int vv = frag_vgpr(kg);           // even -> 8B aligned
      uint2* dst = (uint2*)&lds_a[((mt * 32 + L) * 8) + vv];
      *dst = make_uint2(p0, p1);
      if (kk + 1 < KSTEPS)
        __builtin_prefetch(xp + (size_t)m * II + (k0 + BK + kg), 0, 0);
    }

    // ---- load W: w0,w1 as b128 rows (dominant 256 MB stream), fuse
    //      (w0-w1) bf16 conversion + colsum(w1) bias accumulation ----
#pragma unroll
    for (int h = 0; h < 2; ++h) {
      const int ng = ng0 + 16 * h;            // 0..31 column group
      const int n  = ng * 4;
      const int k  = 2 * sub;                 // even, 0..30
      const f4 a0 = *(const f4*)(w0 + (size_t)(k0 + k)     * OO + n);
      const f4 a1 = *(const f4*)(w0 + (size_t)(k0 + k + 1) * OO + n);
      const f4 b0 = *(const f4*)(w1 + (size_t)(k0 + k)     * OO + n);
      const f4 b1 = *(const f4*)(w1 + (size_t)(k0 + k + 1) * OO + n);
      bias_acc[h] += b0 + b1;
      const int vv = frag_vgpr(k);
#pragma unroll
      for (int c = 0; c < 4; ++c) {
        const int nn = n + c;
        const int L  = frag_lane(nn, k);
        lds_w[(((nn >> 4) * 32 + L) * 8) + vv] =
            pack_bf16(a0[c] - b0[c], a1[c] - b1[c]);
      }
      if (kk + 1 < KSTEPS) {
        __builtin_prefetch(w0 + (size_t)(k0 + BK + k) * OO + n, 0, 0);
        __builtin_prefetch(w1 + (size_t)(k0 + BK + k) * OO + n, 0, 0);
      }
    }

    __syncthreads();

    // ---- WMMA: each wave 2 (M) x 4 (N) 16x16 subtiles, K=32 ----
    v16bf afrag[2], bfrag[4];
#pragma unroll
    for (int mi = 0; mi < 2; ++mi)
      afrag[mi] = *(const v16bf*)&lds_a[(((wm * 2 + mi) * 32 + lane) * 8)];
#pragma unroll
    for (int ni = 0; ni < 4; ++ni)
      bfrag[ni] = *(const v16bf*)&lds_w[(((wn * 4 + ni) * 32 + lane) * 8)];

#pragma unroll
    for (int mi = 0; mi < 2; ++mi)
#pragma unroll
      for (int ni = 0; ni < 4; ++ni)
        acc[mi][ni] = __builtin_amdgcn_wmma_f32_16x16x32_bf16(
            false, afrag[mi], false, bfrag[ni],
            (short)0, acc[mi][ni], false, false);

    __syncthreads();
  }

  // ---- reduce bias partials: column n summed over the 16 kpair-threads ----
#pragma unroll
  for (int h = 0; h < 2; ++h) {
    const int base = (((ng0 + 16 * h) * 16) + sub) * 4;
#pragma unroll
    for (int c = 0; c < 4; ++c)
      lds_bp[base + c] = bias_acc[h][c];
  }
  __syncthreads();
  if (tid < BN) {
    const int ng = tid >> 2, c = tid & 3;
    float s = 0.0f;
#pragma unroll
    for (int sb = 0; sb < 16; ++sb)
      s += lds_bp[((ng * 16) + sb) * 4 + c];
    lds_bias[tid] = s;
  }
  __syncthreads();

  // ---- epilogue: C/D layout (vgpr r -> M=r / M=8+r), fuse bias, store ----
#pragma unroll
  for (int ni = 0; ni < 4; ++ni) {
    const int nt   = wn * 4 + ni;
    const int ncol = nt * 16 + (lane & 15);
    const float bias = lds_bias[ncol];
#pragma unroll
    for (int mi = 0; mi < 2; ++mi) {
      const int mbase = (wm * 2 + mi) * 16 + ((lane >> 4) << 3);
#pragma unroll
      for (int r = 0; r < 8; ++r)
        outp[(size_t)(mbase + r) * OO + ncol] = acc[mi][ni][r] + bias;
    }
  }
}

extern "C" void kernel_launch(void* const* d_in, const int* in_sizes, int n_in,
                              void* d_out, int out_size, void* d_ws, size_t ws_size,
                              hipStream_t stream) {
  (void)in_sizes; (void)n_in; (void)d_ws; (void)ws_size; (void)out_size;
  const float* x = (const float*)d_in[0];   // [P,B,I]
  const float* w = (const float*)d_in[1];   // [2,P,I,O]
  float* out = (float*)d_out;               // [P,B,O]

  // x-dim = 32 tiles of one p (8 MB w + 2 MB x per p fits the 192 MB L2),
  // y-dim = p (slowest) so L2 absorbs intra-p tile re-reads.
  dim3 grid(MTILES * NTILES, PP);
  dim3 block(256);
  evo_binarized_wmma<<<grid, block, 0, stream>>>(x, w, out);
}
